// GPTQLinear_20675972563266
// MI455X (gfx1250) — compile-verified, matrix operand order checked by reading, not run on hardware
//
#include <hip/hip_runtime.h>

#define IN_F     4096
#define OUT_F    4096
#define GROUP    128
#define N_GROUPS 32

typedef __attribute__((ext_vector_type(16))) _Float16 v16h;
typedef __attribute__((ext_vector_type(8)))  _Float16 v8h;
typedef __attribute__((ext_vector_type(4)))  _Float16 v4h;
typedef __attribute__((ext_vector_type(8)))  float    v8f;

// ---- GEMM tiling ----
#define BM 128
#define BN 128
#define BK 128                    // == GROUP, so scale boundary aligns with K-block
#define PITCH_B 272               // LDS row pitch bytes: 128 fp16 + 8 fp16 pad (bank-conflict free)
#define TILE_LDS (BM * PITCH_B)   // 34816 B per matrix per stage
#define STAGE_LDS (2 * TILE_LDS)  // A + B
#define TOTAL_LDS (2 * STAGE_LDS) // double buffered = 139264 B (fits 320KB WGP LDS)

// ---------------- prep: x[:, perm] -> fp16 ----------------
__global__ __launch_bounds__(256) void gptq_prep_x(
    const float* __restrict__ x, const long long* __restrict__ perm,
    _Float16* __restrict__ xh, int rows)
{
  long long idx = (long long)blockIdx.x * 256 + threadIdx.x;
  int k = (int)(idx & (IN_F - 1));
  int n = (int)(idx >> 12);
  if (n >= rows) return;
  int p = (int)perm[k];
  xh[(size_t)n * IN_F + k] = (_Float16)x[(size_t)n * IN_F + p];
}

// ---------------- prep: w_q int32 -> fp16 (exact: |w|<=128) ----------------
__global__ __launch_bounds__(256) void gptq_prep_w(
    const int* __restrict__ wq, _Float16* __restrict__ wh)
{
  size_t i = ((size_t)blockIdx.x * 256 + threadIdx.x) * 4;
  int4 q = *(const int4*)(wq + i);
  v4h h;
  h.x = (_Float16)(float)q.x;
  h.y = (_Float16)(float)q.y;
  h.z = (_Float16)(float)q.z;
  h.w = (_Float16)(float)q.w;
  *(v4h*)(wh + i) = h;
}

// ---------------- CDNA5 async copy helpers ----------------
__device__ __forceinline__ void async_copy_b128(unsigned lds_off, const void* sbase, int goff)
{
  // GVS mode: global address is 64-bit SGPR base plus 32-bit VGPR offset; LDS address comes from the vdst VGPR
  asm volatile("global_load_async_to_lds_b128 %0, %1, %2"
               :: "v"(lds_off), "v"(goff), "s"(sbase)
               : "memory");
}
__device__ __forceinline__ void wait_async0()
{
  asm volatile("s_wait_asynccnt 0" ::: "memory");
}

// ---------------- WMMA GEMM: out = (xh @ wh^T) * s_w (grouped) + bias ----------------
__global__ __launch_bounds__(256, 1) void gptq_wmma_gemm(
    const _Float16* __restrict__ xh,   // rows x IN_F, row major
    const _Float16* __restrict__ wh,   // OUT_F x IN_F, row major (K contiguous)
    const float* __restrict__ s_w,     // N_GROUPS x OUT_F
    const float* __restrict__ bias,    // OUT_F
    float* __restrict__ out,           // rows x OUT_F
    int rows)
{
  extern __shared__ char smem[];
  // low 32 bits of a flat shared pointer == LDS byte address (ISA 10.2 LDS aperture)
  const unsigned lds_base = (unsigned)(size_t)(void*)smem;

  const int tid  = threadIdx.x;     // 0..255, 8 waves
  const int lane = tid & 31;
  const int wid  = tid >> 5;
  const int wm   = wid & 3;         // 4 waves along M  -> rows wm*32..+32
  const int wn   = wid >> 2;        // 2 waves along N  -> cols wn*64..+64
  const int half = lane >> 4;       // K-half select within fragment
  const int l16  = lane & 15;

  const int bm = blockIdx.y * BM;
  const int bn = blockIdx.x * BN;

  v8f accg[2][4];   // per-group f32 accumulators (2 m-tiles x 4 n-tiles)
  v8f fin[2][4];    // scaled running result
#pragma unroll
  for (int i = 0; i < 2; ++i)
#pragma unroll
    for (int j = 0; j < 4; ++j) { v8f z = {}; accg[i][j] = z; fin[i][j] = z; }

  // cooperative stage loader: 256 lanes x 8 rounds x 16B cover one 128x128 fp16 tile
  auto load_stage = [&](int kb, int buf) {
    const void* ab = (const void*)(xh + (size_t)bm * IN_F + (size_t)kb * BK);
    const void* bb = (const void*)(wh + (size_t)bn * IN_F + (size_t)kb * BK);
    const unsigned la = lds_base + (unsigned)(buf * STAGE_LDS);
    const unsigned lb = la + TILE_LDS;
#pragma unroll
    for (int r = 0; r < 8; ++r) {
      int c   = r * 256 + tid;
      int row = c >> 4;                 // 0..127
      int col = (c & 15) * 16;          // byte within 256B row payload
      int go  = row * (IN_F * 2) + col; // global byte offset within tile
      unsigned lo = (unsigned)(row * PITCH_B + col);
      async_copy_b128(la + lo, ab, go);
      async_copy_b128(lb + lo, bb, go);
    }
  };

  load_stage(0, 0);
  wait_async0();
  __syncthreads();

  for (int kb = 0; kb < N_GROUPS; ++kb) {
    const int cur = kb & 1;
    if (kb + 1 < N_GROUPS) load_stage(kb + 1, cur ^ 1);

    const unsigned abase = (unsigned)(cur * STAGE_LDS);
    const unsigned bbase = abase + TILE_LDS;

#pragma unroll
    for (int kk = 0; kk < 4; ++kk) {            // 4 x K=32 per group
      const unsigned kbyte = (unsigned)(kk * 64 + half * 16);

      v16h afrag[2], bfrag[4];
#pragma unroll
      for (int tm = 0; tm < 2; ++tm) {
        unsigned o = abase + (unsigned)((wm * 32 + tm * 16 + l16) * PITCH_B) + kbyte;
        v8h lo = *(const v8h*)(smem + o);
        v8h hi = *(const v8h*)(smem + o + 32);
        afrag[tm] = __builtin_shufflevector(lo, hi, 0,1,2,3,4,5,6,7,8,9,10,11,12,13,14,15);
      }
#pragma unroll
      for (int tn = 0; tn < 4; ++tn) {
        unsigned o = bbase + (unsigned)((wn * 64 + tn * 16 + l16) * PITCH_B) + kbyte;
        v8h lo = *(const v8h*)(smem + o);
        v8h hi = *(const v8h*)(smem + o + 32);
        bfrag[tn] = __builtin_shufflevector(lo, hi, 0,1,2,3,4,5,6,7,8,9,10,11,12,13,14,15);
      }
#pragma unroll
      for (int tm = 0; tm < 2; ++tm)
#pragma unroll
        for (int tn = 0; tn < 4; ++tn)
          accg[tm][tn] = __builtin_amdgcn_wmma_f32_16x16x32_f16(
              false, afrag[tm], false, bfrag[tn],
              (short)0, accg[tm][tn], false, false);
    }

    // fold per-(group, out) scale into running f32 result
#pragma unroll
    for (int tn = 0; tn < 4; ++tn) {
      float sc = s_w[(size_t)kb * OUT_F + bn + wn * 64 + tn * 16 + l16];
#pragma unroll
      for (int tm = 0; tm < 2; ++tm) {
        fin[tm][tn] += accg[tm][tn] * sc;
        v8f z = {}; accg[tm][tn] = z;
      }
    }

    if (kb + 1 < N_GROUPS) { wait_async0(); __syncthreads(); }
  }

  // epilogue: bias + store (C/D layout: VGPR v -> M = v + half*8, lane%16 -> N)
#pragma unroll
  for (int tn = 0; tn < 4; ++tn) {
    int coln = bn + wn * 64 + tn * 16 + l16;
    float bs = bias[coln];
#pragma unroll
    for (int tm = 0; tm < 2; ++tm) {
      int row0 = bm + wm * 32 + tm * 16 + half * 8;
      v8f r = fin[tm][tn];
#pragma unroll
      for (int v = 0; v < 8; ++v)
        out[(size_t)(row0 + v) * OUT_F + coln] = r[v] + bs;
    }
  }
}

extern "C" void kernel_launch(void* const* d_in, const int* in_sizes, int n_in,
                              void* d_out, int out_size, void* d_ws, size_t ws_size,
                              hipStream_t stream)
{
  const float*     x    = (const float*)d_in[0];
  const int*       wq   = (const int*)d_in[1];
  const float*     sw   = (const float*)d_in[2];
  const long long* perm = (const long long*)d_in[3];
  const float*     bias = (const float*)d_in[4];
  float*           out  = (float*)d_out;

  const int rows = in_sizes[0] / IN_F;   // 2048 for the reference shapes

  _Float16* xh = (_Float16*)d_ws;
  size_t xh_bytes = (((size_t)rows * IN_F * sizeof(_Float16)) + 255) & ~(size_t)255;
  _Float16* wh = (_Float16*)((char*)d_ws + xh_bytes);

  {
    long long tot = (long long)rows * IN_F;
    int blocks = (int)((tot + 255) / 256);
    gptq_prep_x<<<blocks, 256, 0, stream>>>(x, perm, xh, rows);
  }
  {
    size_t tot = (size_t)OUT_F * IN_F / 4;
    int blocks = (int)((tot + 255) / 256);
    gptq_prep_w<<<blocks, 256, 0, stream>>>(wq, wh);
  }

  dim3 grid(OUT_F / BN, rows / BM);
  gptq_wmma_gemm<<<grid, 256, TOTAL_LDS, stream>>>(xh, wh, sw, bias, out, rows);
}
